// EGSC_ScoringHead_14620068676118
// MI455X (gfx1250) — compile-verified, matrix operand order checked by reading, not run on hardware
//
#include <hip/hip_runtime.h>
#include <hip/hip_bf16.h>

typedef _Float16 v16h __attribute__((ext_vector_type(16)));
typedef _Float16 v8h  __attribute__((ext_vector_type(8)));
typedef _Float16 v4h  __attribute__((ext_vector_type(4)));
typedef float    v8f  __attribute__((ext_vector_type(8)));

// ---------------- problem constants (fixed by the reference) ----------------
constexpr int kN     = 524288;   // nodes
constexpr int kD     = 256;      // feature dim
constexpr int kH     = 64;       // D / R
constexpr int kG     = 2048;     // graphs
constexpr int kPairs = kG / 2;   // 1024
constexpr int kNW    = 8;                         // waves per block (wave32)
constexpr int kTiles = kN / 16;                   // 32768 row tiles
constexpr int kNodeBlocks   = 512;
constexpr int kTilesPerWave = kTiles / (kNodeBlocks * kNW); // 8

// ---------------- WMMA helpers ----------------
__device__ __forceinline__ v8f wmma_f16(v16h a, v16h b, v8f c) {
  // D = A(16x32 f16) * B(32x16 f16) + C(16x16 f32)
  return __builtin_amdgcn_wmma_f32_16x16x32_f16(false, a, false, b, (short)0, c,
                                                false, false);
}

// A fragment from a row-major f16 LDS tile (16 rows, strideF16 columns).
// Per ISA layout: lane r=l&15 owns row r; lanes 0-15 take K pairs {0..7,16..23},
// lanes 16-31 take {8..15,24..31} -> two contiguous 16B chunks per lane.
__device__ __forceinline__ v16h ldsA(const _Float16* base, int strideF16, int kc,
                                     int lane) {
  const int r    = lane & 15;
  const int koff = (lane >> 4) << 3;
  const _Float16* p = base + r * strideF16 + kc * 32 + koff;
  union { v16h v; v8h h[2]; } u;
  u.h[0] = *(const v8h*)(p);
  u.h[1] = *(const v8h*)(p + 16);
  return u.v;
}

// B fragment from a pre-swizzled LDS image: fragment (kc,nt) is 32 lanes x 16 f16,
// each lane's 32 bytes contiguous.
__device__ __forceinline__ v16h ldsB(const _Float16* sW, int NT, int kc, int nt,
                                     int lane) {
  const _Float16* p = sW + ((size_t)((kc * NT + nt) * 32 + lane)) * 16;
  union { v16h v; v8h h[2]; } u;
  u.h[0] = *(const v8h*)(p);
  u.h[1] = *(const v8h*)(p + 8);
  return u.v;
}

// Convert a row-major fp32 KxNn weight matrix to f16 B fragments in LDS.
// Inverse of the B layout assumed by ldsB (mirrors the documented A layout with
// N taking the role of M).
__device__ void swizzleB(const float* __restrict__ W, _Float16* __restrict__ dst,
                         int K, int Nn, int tid, int nthreads) {
  const int NT = Nn >> 4;
  for (int idx = tid; idx < K * Nn; idx += nthreads) {
    const int k = idx / Nn, n = idx - k * Nn;
    const int kc = k >> 5, kr = k & 31;
    const int c = (kr >> 4) & 1, t = kr & 15;
    const int hi = t >> 3, j = t & 7;
    const int lane = hi * 16 + (n & 15);
    const int nt = n >> 4;
    dst[((size_t)((kc * NT + nt) * 32 + lane)) * 16 + c * 8 + j] =
        (_Float16)W[idx];
  }
}

// ---------------- zero workspace ----------------
__global__ void zero_kernel(float* __restrict__ p, int n) {
  for (int i = blockIdx.x * blockDim.x + threadIdx.x; i < n;
       i += gridDim.x * blockDim.x)
    p[i] = 0.f;
}

// ---------------- node stage ----------------
// MODE 0: gate MLP -> scaled; segmented accumulate into sums / counts.
// MODE 1: recompute scaled; coef = sigmoid(scaled . ctx[g]); segmented
//         accumulate coef*scaled into pooled.
template <int MODE>
__global__ __launch_bounds__(kNW * 32) void node_kernel(
    const float* __restrict__ X, const float* __restrict__ W1,
    const float* __restrict__ B1, const float* __restrict__ W2,
    const float* __restrict__ B2, const int* __restrict__ gidx,
    const float* __restrict__ ctx, float* __restrict__ sums,
    float* __restrict__ counts, float* __restrict__ pooled) {
  extern __shared__ char smem_raw[];
  _Float16* sW1 = (_Float16*)smem_raw;            // 8*4*512   = 16384 f16
  _Float16* sW2 = sW1 + 16384;                    // 2*16*512  = 16384 f16
  _Float16* XT  = sW2 + 16384;                    // kNW * 16*256 f16
  _Float16* HT  = XT + kNW * 4096;                // kNW * 16*64  f16
  float* sB1   = (float*)(HT + kNW * 1024);       // 64
  float* sB2   = sB1 + kH;                        // 256
  float* sCoef = sB2 + kD;                        // kNW*16
  int*   sGix  = (int*)(sCoef + kNW * 16);        // kNW*16

  const int tid  = threadIdx.x;
  const int wave = tid >> 5;
  const int lane = tid & 31;

  swizzleB(W1, sW1, kD, kH, tid, blockDim.x);
  swizzleB(W2, sW2, kH, kD, tid, blockDim.x);
  for (int i = tid; i < kH; i += blockDim.x) sB1[i] = B1[i];
  for (int i = tid; i < kD; i += blockDim.x) sB2[i] = B2[i];
  __syncthreads();

  _Float16* myX = XT + wave * 4096;   // 16 x 256 f16 tile (becomes `scaled`)
  _Float16* myH = HT + wave * 1024;   // 16 x 64  f16 hidden tile
  int*   myG = sGix + wave * 16;
  float* myC = sCoef + wave * 16;

  // per-lane running segment accumulators: columns lane, lane+32, ..., lane+224
  float colAcc[8];
#pragma unroll
  for (int i = 0; i < 8; ++i) colAcc[i] = 0.f;
  float cntAcc = 0.f;
  int curg = -1;

  const int n16 = lane & 15, hi = lane >> 4;
  const int tile0 = (blockIdx.x * kNW + wave) * kTilesPerWave;

  for (int t = 0; t < kTilesPerWave; ++t) {
    const int node0 = (tile0 + t) * 16;
    if (lane < 16) myG[lane] = gidx[node0 + lane];

    // load 16x256 fp32 tile coalesced, convert to f16 in LDS
    const float4* Xg = (const float4*)(X + (size_t)node0 * kD);
    for (int i = lane; i < 1024; i += 32) {
      float4 xv = Xg[i];
      const int row = i >> 6, c4 = (i & 63) << 2;
      v4h h4 = {(_Float16)xv.x, (_Float16)xv.y, (_Float16)xv.z, (_Float16)xv.w};
      *(v4h*)(myX + row * kD + c4) = h4;
    }

    // prefetch next tile's 16KB of X (one 128B line per lane per step) so the
    // HBM pipe stays full while this tile runs through the WMMA stages
    if (t + 1 < kTilesPerWave) {
      const char* nx = (const char*)(X + (size_t)(node0 + 16) * kD);
#pragma unroll
      for (int pf = 0; pf < 4; ++pf)
        __builtin_prefetch(nx + pf * 4096 + lane * 128, 0, 3);
    }
    __syncthreads();

    // ---- stage 1: hidden = relu(X @ W1 + b1), 16x64 ----
    v8f acc1[4];
#pragma unroll
    for (int nt = 0; nt < 4; ++nt) acc1[nt] = {};
    for (int kc = 0; kc < 8; ++kc) {
      v16h a = ldsA(myX, kD, kc, lane);
#pragma unroll
      for (int nt = 0; nt < 4; ++nt) {
        v16h b = ldsB(sW1, 4, kc, nt, lane);
        acc1[nt] = wmma_f16(a, b, acc1[nt]);
      }
    }
#pragma unroll
    for (int nt = 0; nt < 4; ++nt)
#pragma unroll
      for (int v = 0; v < 8; ++v) {
        const int m = v + 8 * hi, hcol = nt * 16 + n16;
        float val = acc1[nt][v] + sB1[hcol];
        myH[m * kH + hcol] = (_Float16)fmaxf(val, 0.f);
      }
    __syncthreads();

    // ---- stage 2: gate = tanh(hidden @ W2 + b2); scaled overwrites X tile ----
    const v16h a0 = ldsA(myH, kH, 0, lane);
    const v16h a1 = ldsA(myH, kH, 1, lane);
    for (int nt2 = 0; nt2 < 16; ++nt2) {
      v8f acc2 = {};
      acc2 = wmma_f16(a0, ldsB(sW2, 16, 0, nt2, lane), acc2);
      acc2 = wmma_f16(a1, ldsB(sW2, 16, 1, nt2, lane), acc2);
#pragma unroll
      for (int v = 0; v < 8; ++v) {
        const int m = v + 8 * hi, col = nt2 * 16 + n16;
        const float gate = tanhf(acc2[v] + sB2[col]);
        const float sc = (gate + 1.f) * (float)myX[m * kD + col];
        myX[m * kD + col] = (_Float16)sc;
      }
    }
    __syncthreads();

    if (MODE == 1) {
      // per-row dot(scaled, ctx[g]) -> coef
      const int r = n16, half = hi;
      const float* cr = ctx + (size_t)myG[r] * kD + half * 128;
      const _Float16* sr = myX + r * kD + half * 128;
      float dot = 0.f;
      for (int c = 0; c < 128; ++c) dot = fmaf((float)sr[c], cr[c], dot);
      dot += __shfl_xor(dot, 16, 32);
      if (lane < 16) myC[r] = 1.f / (1.f + expf(-dot));
      __syncthreads();
    }

    // ---- segmented accumulation over sorted graph ids ----
    for (int r = 0; r < 16; ++r) {
      const int g = myG[r];                    // uniform across the wave
      if (g != curg) {
        if (curg >= 0) {
          float* dst = (MODE == 0) ? sums : pooled;
#pragma unroll
          for (int i = 0; i < 8; ++i) {
            atomicAdd(dst + (size_t)curg * kD + lane + 32 * i, colAcc[i]);
            colAcc[i] = 0.f;
          }
          if (MODE == 0 && lane == 0) atomicAdd(&counts[curg], cntAcc);
          cntAcc = 0.f;
        }
        curg = g;
      }
      const float w = (MODE == 0) ? 1.f : myC[r];
#pragma unroll
      for (int i = 0; i < 8; ++i)
        colAcc[i] += w * (float)myX[r * kD + lane + 32 * i];
      if (MODE == 0 && lane == 0) cntAcc += 1.f;
    }
    __syncthreads();
  }

  if (curg >= 0) {
    float* dst = (MODE == 0) ? sums : pooled;
#pragma unroll
    for (int i = 0; i < 8; ++i)
      atomicAdd(dst + (size_t)curg * kD + lane + 32 * i, colAcc[i]);
    if (MODE == 0 && lane == 0) atomicAdd(&counts[curg], cntAcc);
  }
}

// ---------------- graph context: ctx = tanh((sums/counts) @ p_mt) ----------------
__global__ __launch_bounds__(256) void ctx_kernel(const float* __restrict__ sums,
                                                  const float* __restrict__ counts,
                                                  const float* __restrict__ Pmt,
                                                  float* __restrict__ ctx) {
  __shared__ float mean[kD];
  const int g = blockIdx.x;
  const float c = counts[g];
  const float inv = (c > 0.f) ? 1.f / c : 0.f;
  for (int i = threadIdx.x; i < kD; i += blockDim.x)
    mean[i] = sums[(size_t)g * kD + i] * inv;
  __syncthreads();
  const int n = threadIdx.x;
  float acc = 0.f;
  for (int k = 0; k < kD; ++k) acc = fmaf(mean[k], Pmt[k * kD + n], acc);
  ctx[(size_t)g * kD + n] = tanhf(acc);
}

// ---------------- pair head ----------------
// 16 pair-rows per block; out[r][n] loops k with w[k][n] read once and reused
// for all 16 rows (weights stream from L2 once per block).
__device__ void tgemm(const float* __restrict__ W, const float* __restrict__ bias,
                      const float* __restrict__ in, int Nin,
                      float* __restrict__ outp, int Nout, bool relu) {
  for (int n = threadIdx.x; n < Nout; n += blockDim.x) {
    float acc[16];
    const float b = bias ? bias[n] : 0.f;
#pragma unroll
    for (int r = 0; r < 16; ++r) acc[r] = b;
    for (int k = 0; k < Nin; ++k) {
      const float w = W[(size_t)k * Nout + n];
#pragma unroll
      for (int r = 0; r < 16; ++r) acc[r] = fmaf(in[r * 512 + k], w, acc[r]);
    }
#pragma unroll
    for (int r = 0; r < 16; ++r)
      outp[r * 512 + n] = relu ? fmaxf(acc[r], 0.f) : acc[r];
  }
  __syncthreads();
}

__global__ __launch_bounds__(256) void pair_kernel(
    const float* __restrict__ pooled, const float* ig_w1, const float* ig_b1,
    const float* ig_w2, const float* ig_b2, const float* if_w1,
    const float* if_b1, const float* if_w2, const float* if_b2,
    const float* sg_w1, const float* sg_b1, const float* sg_w2,
    const float* sg_b2, const float* sf_w1, const float* sf_b1,
    const float* sf_w2, const float* sf_b2, float* __restrict__ out) {
  extern __shared__ float psm[];
  float* C0 = psm;              // 16 x 512
  float* C1 = C0 + 16 * 512;    // 16 x 512
  float* C2 = C1 + 16 * 512;    // 16 x 512
  const int p0 = blockIdx.x * 16;

  // comb = [pooled[2p], pooled[2p+1]]
  for (int i = threadIdx.x; i < 16 * 512; i += blockDim.x) {
    const int r = i >> 9, c = i & 511;
    const int pr = p0 + r;
    const size_t src =
        (c < 256) ? (size_t)(2 * pr) * kD + c : (size_t)(2 * pr + 1) * kD + (c - 256);
    C0[r * 512 + c] = pooled[src];
  }
  __syncthreads();

  tgemm(ig_w1, ig_b1, C0, 512, C1, 128, true);
  tgemm(ig_w2, ig_b2, C1, 128, C2, 512, false);
  for (int i = threadIdx.x; i < 16 * 512; i += blockDim.x) {
    const int r = i >> 9, c = i & 511;
    const float s = 1.f / (1.f + expf(-C2[r * 512 + c]));
    C0[r * 512 + c] *= (s + 1.f);
  }
  __syncthreads();
  tgemm(if_w1, if_b1, C0, 512, C1, 512, true);
  tgemm(if_w2, if_b2, C1, 512, C2, 256, true);   // inter
  tgemm(sg_w1, sg_b1, C2, 256, C1, 64, true);
  tgemm(sg_w2, sg_b2, C1, 64, C0, 256, false);   // sgate raw
  for (int i = threadIdx.x; i < 16 * 256; i += blockDim.x) {
    const int r = i >> 8, c = i & 255;
    const float s = 1.f / (1.f + expf(-C0[r * 512 + c]));
    C0[r * 512 + c] = (s + 1.f) * C2[r * 512 + c];  // sscaled
  }
  __syncthreads();
  tgemm(sf_w1, sf_b1, C0, 256, C1, 128, true);
  if (threadIdx.x < 16) {
    float acc = sf_b2[0];
    for (int k = 0; k < 128; ++k)
      acc = fmaf(C1[threadIdx.x * 512 + k], sf_w2[k], acc);
    out[p0 + threadIdx.x] = acc;
  }
}

// ---------------- launch ----------------
extern "C" void kernel_launch(void* const* d_in, const int* in_sizes, int n_in,
                              void* d_out, int out_size, void* d_ws,
                              size_t ws_size, hipStream_t stream) {
  const float* X     = (const float*)d_in[0];
  const float* pg_w1 = (const float*)d_in[1];
  const float* pg_b1 = (const float*)d_in[2];
  const float* pg_w2 = (const float*)d_in[3];
  const float* pg_b2 = (const float*)d_in[4];
  const float* p_mt  = (const float*)d_in[5];
  const float* ig_w1 = (const float*)d_in[6];
  const float* ig_b1 = (const float*)d_in[7];
  const float* ig_w2 = (const float*)d_in[8];
  const float* ig_b2 = (const float*)d_in[9];
  const float* if_w1 = (const float*)d_in[10];
  const float* if_b1 = (const float*)d_in[11];
  const float* if_w2 = (const float*)d_in[12];
  const float* if_b2 = (const float*)d_in[13];
  const float* sg_w1 = (const float*)d_in[14];
  const float* sg_b1 = (const float*)d_in[15];
  const float* sg_w2 = (const float*)d_in[16];
  const float* sg_b2 = (const float*)d_in[17];
  const float* sf_w1 = (const float*)d_in[18];
  const float* sf_b1 = (const float*)d_in[19];
  const float* sf_w2 = (const float*)d_in[20];
  const float* sf_b2 = (const float*)d_in[21];
  const int*   gidx  = (const int*)d_in[22];

  float* sums   = (float*)d_ws;            // kG*kD
  float* counts = sums + (size_t)kG * kD;  // kG
  float* pooled = counts + kG;             // kG*kD
  float* ctx    = pooled + (size_t)kG * kD;// kG*kD

  const int zeroN = 2 * kG * kD + kG;
  zero_kernel<<<1024, 256, 0, stream>>>(sums, zeroN);

  const size_t nodeSmem =
      (size_t)(16384 + 16384 + kNW * 4096 + kNW * 1024) * sizeof(_Float16) +
      (size_t)(kH + kD + kNW * 16) * sizeof(float) +
      (size_t)(kNW * 16) * sizeof(int);

  node_kernel<0><<<kNodeBlocks, kNW * 32, nodeSmem, stream>>>(
      X, pg_w1, pg_b1, pg_w2, pg_b2, gidx, nullptr, sums, counts, pooled);

  ctx_kernel<<<kG, 256, 0, stream>>>(sums, counts, p_mt, ctx);

  node_kernel<1><<<kNodeBlocks, kNW * 32, nodeSmem, stream>>>(
      X, pg_w1, pg_b1, pg_w2, pg_b2, gidx, ctx, sums, counts, pooled);

  const size_t pairSmem = (size_t)3 * 16 * 512 * sizeof(float);
  pair_kernel<<<kPairs / 16, 256, pairSmem, stream>>>(
      pooled, ig_w1, ig_b1, ig_w2, ig_b2, if_w1, if_b1, if_w2, if_b2, sg_w1,
      sg_b1, sg_w2, sg_b2, sf_w1, sf_b1, sf_w2, sf_b2, (float*)d_out);
}
